// SelfAttention_37804302139594
// MI455X (gfx1250) — matrix operand order repaired and checked by reading, not compiled, not run on hardware
//
#include <hip/hip_runtime.h>

// ---------------------------------------------------------------------------
// Self-attention forward for MI455X (gfx1250, wave32, WMMA, async-to-LDS).
//   hidden: [4, 2048, 1024] f32 ; Wq/Wk/Wv: [1024,1024] f32 ; b*: [1024] f32
//   out = concat(ctx [4,2048,1024] f32, probs [4,16,2048,2048] f32)
// Strategy:
//  - bf16 WMMA (fp32 accumulate) for all three GEMM stages.
//  - Q/K/V workspace is bf16 (same numerics as converting at consume time,
//    half the HBM traffic); kernel 2's K staging is then a pure async copy.
//  - QKV GEMM: 128x64 tiles, double-buffered fp32 LDS tiles filled with
//    GLOBAL_LOAD_ASYNC_TO_LDS_B128 (ASYNCcnt), overlapping fetch with WMMA.
//  - Attention: 16-query panel per block; 16x2048 fp32 score panel lives in
//    LDS (CDNA5 320KB/WGP) so softmax + probs@V never round-trip HBM; only
//    the mandatory 1.07GB probs write goes out (~46us at 23.3 TB/s).
//  - fp32->bf16 pairs via vector truncate -> single V_CVT_PK_BF16_F32.
// ---------------------------------------------------------------------------

typedef __bf16 bf16x16 __attribute__((ext_vector_type(16)));
typedef __bf16 bf16x2  __attribute__((ext_vector_type(2)));
typedef float  f32x8   __attribute__((ext_vector_type(8)));
typedef float  f32x4   __attribute__((ext_vector_type(4)));
typedef float  f32x2   __attribute__((ext_vector_type(2)));
typedef int    i32x4   __attribute__((ext_vector_type(4)));
typedef unsigned int u32x4 __attribute__((ext_vector_type(4)));

#if __has_builtin(__builtin_amdgcn_global_load_async_to_lds_b128) && \
    __has_builtin(__builtin_amdgcn_s_wait_asynccnt)
#define HAVE_ASYNC 1
#define WAIT_ASYNC(n) __builtin_amdgcn_s_wait_asynccnt(n)
#else
#define HAVE_ASYNC 0
#define WAIT_ASYNC(n) ((void)0)
#endif

union Frag {
  bf16x16        v;
  unsigned short u[16];
  unsigned int   w[8];
  u32x4          q[2];
};
union Pack8 {                       // 8 bf16 loaded as one b128
  u32x4          q;
  unsigned short u[8];
};

// fp32 -> bf16 (native conversion; single-element form).
__device__ __forceinline__ unsigned short f2bf(float f) {
  union { __bf16 h; unsigned short u; } t;
  t.h = (__bf16)f;
  return t.u;
}
// Two fp32 -> packed bf16 pair; vector truncate lowers to V_CVT_PK_BF16_F32.
__device__ __forceinline__ unsigned int pack2(float a, float b) {
  f32x2 s = {a, b};
  union { bf16x2 h; unsigned int u; } t;
  t.h = __builtin_convertvector(s, bf16x2);
  return t.u;
}
__device__ __forceinline__ f32x8 wmma_bf16(const Frag& a, const Frag& b, f32x8 c) {
  return __builtin_amdgcn_wmma_f32_16x16x32_bf16(false, a.v, false, b.v,
                                                 (short)0, c, false, false);
}

// 16B global -> LDS copy (async on CDNA5, sync fallback otherwise).
// Builtin signature (probe-confirmed): (int4 AS1* src, int4 AS3* lds,
// imm offset, imm cpol).
__device__ __forceinline__ void async_b128(const void* g, void* l) {
#if HAVE_ASYNC
  __builtin_amdgcn_global_load_async_to_lds_b128(
      (__attribute__((address_space(1))) i32x4*)g,
      (__attribute__((address_space(3))) i32x4*)l, 0, 0);
#else
  *(u32x4*)l = *(const u32x4*)g;
#endif
}

__device__ __forceinline__ void cvt4(Frag& f, int base, f32x4 v) {
  f.w[base / 2]     = pack2(v[0], v[1]);
  f.w[base / 2 + 1] = pack2(v[2], v[3]);
}
// A fragment from a fp32 LDS row (k-window base `row`):
//   elem i -> k = 16g+i
__device__ __forceinline__ Frag fragA_f32(const float* row, int g) {
  Frag f;
  cvt4(f, 0,  *(const f32x4*)(row + 16 * g));
  cvt4(f, 4,  *(const f32x4*)(row + 16 * g + 4));
  cvt4(f, 8,  *(const f32x4*)(row + 16 * g + 8));
  cvt4(f, 12, *(const f32x4*)(row + 16 * g + 12));
  return f;
}
// B fragment from a fp32 LDS row pre-offset by +16g (elem i -> k = 16g+i).
__device__ __forceinline__ Frag fragB_f32(const float* p) {
  Frag f;
  cvt4(f, 0,  *(const f32x4*)(p));
  cvt4(f, 4,  *(const f32x4*)(p + 4));
  cvt4(f, 8,  *(const f32x4*)(p + 8));
  cvt4(f, 12, *(const f32x4*)(p + 12));
  return f;
}
// A fragment straight from a bf16 LDS row (k-window base `row`).
__device__ __forceinline__ Frag fragA_bf16(const unsigned short* row, int g) {
  Frag f;
  f.q[0] = *(const u32x4*)(row + 16 * g);
  f.q[1] = *(const u32x4*)(row + 16 * g + 8);
  return f;
}
// B fragment straight from a bf16 LDS row pre-offset by +16g.
__device__ __forceinline__ Frag fragB_bf16(const unsigned short* p) {
  Frag f;
  f.q[0] = *(const u32x4*)(p);
  f.q[1] = *(const u32x4*)(p + 8);
  return f;
}

// ---------------------------------------------------------------------------
// Kernel 1: y = x @ W^T + bias -> bf16, head-split layout [B,H,S,64].
// grid = (N/64, M/128, 3{q,k,v}), block = 256 (8 waves), tile 128x64, K-step 32.
// Double-buffered async-to-LDS staging (6 async b128 per thread per k-step).
// ---------------------------------------------------------------------------
__device__ __forceinline__ void stage_k1(const float* __restrict__ hsB,
                                         const float* __restrict__ WB,
                                         int kbase, float* aDst, float* bDst,
                                         int tid) {
#pragma unroll
  for (int i = 0; i < 4; ++i) {                    // A: 128 x 32 fp32
    int idx = tid + 256 * i, row = idx >> 3, c4 = (idx & 7) * 4;
    async_b128(&hsB[(size_t)row * 1024 + kbase + c4], &aDst[row * 32 + c4]);
  }
#pragma unroll
  for (int i = 0; i < 2; ++i) {                    // B: 64 x 32 fp32 (W rows)
    int idx = tid + 256 * i, row = idx >> 3, c4 = (idx & 7) * 4;
    async_b128(&WB[(size_t)row * 1024 + kbase + c4], &bDst[row * 32 + c4]);
  }
}

__global__ __launch_bounds__(256) void qkv_proj_kernel(
    const float* __restrict__ hs,
    const float* __restrict__ Wq, const float* __restrict__ bq,
    const float* __restrict__ Wk, const float* __restrict__ bk,
    const float* __restrict__ Wv, const float* __restrict__ bv,
    unsigned short* __restrict__ Q, unsigned short* __restrict__ K,
    unsigned short* __restrict__ V)
{
  __shared__ __align__(16) float aT[2][128 * 32];  // 16KB x2
  __shared__ __align__(16) float bT[2][64 * 32];   //  8KB x2

  const int z = blockIdx.z;
  const float* W    = (z == 0) ? Wq : (z == 1) ? Wk : Wv;
  const float* bias = (z == 0) ? bq : (z == 1) ? bk : bv;
  unsigned short* out = (z == 0) ? Q : (z == 1) ? K : V;

  const int n0 = blockIdx.x * 64;
  const int m0 = blockIdx.y * 128;
  const int tid = threadIdx.x;
  const int lane = tid & 31, wave = tid >> 5;
  const int g = lane >> 4, ln = lane & 15;
  const int wm = (wave >> 1) * 32, wn = (wave & 1) * 32;

  const float* hsB = hs + (size_t)m0 * 1024;
  const float* WB  = W  + (size_t)n0 * 1024;

  f32x8 acc[2][2] = {};

  stage_k1(hsB, WB, 0, aT[0], bT[0], tid);
  for (int kt = 0; kt < 32; ++kt) {
    const int cur = kt & 1;
    if (kt < 31) {
      stage_k1(hsB, WB, (kt + 1) * 32, aT[1 - cur], bT[1 - cur], tid);
      WAIT_ASYNC(6);          // oldest 6 (this k-step's tile) have landed
    } else {
      WAIT_ASYNC(0);
    }
    __syncthreads();          // all waves' async stores for `cur` visible

    Frag A[2], B[2];
#pragma unroll
    for (int i = 0; i < 2; ++i) {
      A[i] = fragA_f32(&aT[cur][(wm + 16 * i + ln) * 32], g);
      B[i] = fragB_f32(&bT[cur][(wn + 16 * i + ln) * 32 + 16 * g]);
    }
#pragma unroll
    for (int i = 0; i < 2; ++i)
#pragma unroll
      for (int j = 0; j < 2; ++j)
        acc[i][j] = wmma_bf16(A[i], B[j], acc[i][j]);
    __syncthreads();          // done reading `cur` before it is re-staged
  }

  // Store bf16: D layout m = 8g + r per VGPR r, n = lane%16. Head-split.
#pragma unroll
  for (int j = 0; j < 2; ++j) {
    int n_glob = n0 + wn + 16 * j + ln;
    float bvl = bias[n_glob];
    int h = n_glob >> 6, dd = n_glob & 63;
#pragma unroll
    for (int i = 0; i < 2; ++i) {
      int mbase = m0 + wm + 16 * i + 8 * g;
#pragma unroll
      for (int r = 0; r < 8; ++r) {
        int m_glob = mbase + r;
        int bb = m_glob >> 11, s = m_glob & 2047;
        out[(((size_t)(bb * 16 + h)) * 2048 + s) * 64 + dd] =
            f2bf(acc[i][j][r] + bvl);
      }
    }
  }
}

// ---------------------------------------------------------------------------
// Kernel 2: per (b, h, 16-query tile): scores -> softmax -> probs + ctx.
// grid = (128, 16, 4), block = 256 (8 waves). LDS ~150KB.
// ---------------------------------------------------------------------------
__global__ __launch_bounds__(256) void attn_kernel(
    const unsigned short* __restrict__ Q, const unsigned short* __restrict__ K,
    const unsigned short* __restrict__ V, const float* __restrict__ mask,
    float* __restrict__ ctx, float* __restrict__ probs)
{
  __shared__ __align__(16) float          sS[16 * 2048];   // score panel (128KB)
  __shared__ __align__(16) unsigned short sKV[128 * 64];   // K chunk / V^T chunk
  __shared__ __align__(16) unsigned short sQ[16 * 64];     // Q tile bf16 (2KB)
  __shared__ __align__(16) float          sCtx[16 * 64];   // ctx accumulator

  const int qt = blockIdx.x, h = blockIdx.y, b = blockIdx.z;
  const int bh = b * 16 + h;
  const int q0 = qt * 16;
  const int tid = threadIdx.x, lane = tid & 31, wave = tid >> 5;
  const int g = lane >> 4, ln = lane & 15;

  const unsigned short* Qp = Q + ((size_t)bh * 2048 + q0) * 64;
  const unsigned short* Kp = K + (size_t)bh * 2048 * 64;
  const unsigned short* Vp = V + (size_t)bh * 2048 * 64;
  const float* maskp = mask + (size_t)b * 2048;

  // Stage Q tile (16 x 64 bf16 = 128 x b128) with async copies.
  if (tid < 128) {
    int row = tid >> 3, c8 = (tid & 7) * 8;
    async_b128(&Qp[row * 64 + c8], &sQ[row * 64 + c8]);
  }
  WAIT_ASYNC(0);
  __syncthreads();

  Frag aq[2];
#pragma unroll
  for (int ks = 0; ks < 2; ++ks)
    aq[ks] = fragA_bf16(&sQ[ln * 64 + ks * 32], g);

  // ---- Phase 1: scores = Q K^T / 8 + mask, into LDS panel ----
  for (int c = 0; c < 16; ++c) {
    const int kk0 = c * 128;
    if (c < 15) __builtin_prefetch(&Kp[(size_t)(kk0 + 128) * 64], 0, 0);
    // Stage K chunk (128 x 64 bf16 = 1024 x b128): pure async copy.
#pragma unroll
    for (int i = 0; i < 4; ++i) {
      int idx = tid + 256 * i, row = idx >> 3, c8 = (idx & 7) * 8;
      async_b128(&Kp[(size_t)(kk0 + row) * 64 + c8], &sKV[row * 64 + c8]);
    }
    WAIT_ASYNC(0);
    __syncthreads();

    const int n0v = wave * 16;               // this wave's 16-key tile
    f32x8 acc = {};
#pragma unroll
    for (int ks = 0; ks < 2; ++ks) {
      Frag bk = fragB_bf16(&sKV[(n0v + ln) * 64 + ks * 32 + 16 * g]);
      acc = wmma_bf16(aq[ks], bk, acc);
    }
    int keyg = kk0 + n0v + ln;
    float madd = maskp[keyg];
#pragma unroll
    for (int r = 0; r < 8; ++r)
      sS[(8 * g + r) * 2048 + keyg] = acc[r] * 0.125f + madd;
    __syncthreads();
  }

  // ---- Phase 2: softmax per row (2 rows per wave), write probs ----
#pragma unroll
  for (int rr = 0; rr < 2; ++rr) {
    int row = wave * 2 + rr;
    float* rowp = &sS[row * 2048];
    float mx = -3.4e38f;
    for (int i = 0; i < 64; ++i) mx = fmaxf(mx, rowp[lane + 32 * i]);
#pragma unroll
    for (int off = 16; off >= 1; off >>= 1) mx = fmaxf(mx, __shfl_xor(mx, off, 32));
    float sum = 0.f;
    for (int i = 0; i < 64; ++i) {
      float e = __expf(rowp[lane + 32 * i] - mx);
      rowp[lane + 32 * i] = e;
      sum += e;
    }
#pragma unroll
    for (int off = 16; off >= 1; off >>= 1) sum += __shfl_xor(sum, off, 32);
    float inv = 1.0f / sum;
    float* gp = &probs[((size_t)bh * 2048 + q0 + row) * 2048];
    for (int i = 0; i < 64; ++i) {
      float p = rowp[lane + 32 * i] * inv;
      rowp[lane + 32 * i] = p;
      gp[lane + 32 * i] = p;      // coalesced probs write (the 1GB output)
    }
  }

#pragma unroll
  for (int i = 0; i < 4; ++i) sCtx[tid + 256 * i] = 0.f;
  __syncthreads();

  // ---- Phase 3: ctx = probs @ V ----
  // wave -> (d-tile = wave&3, key-half = wave>>2 within each 128-key chunk)
  const int ntile = wave & 3;
  const int khalf = wave >> 2;
  f32x8 accc = {};
  for (int c = 0; c < 16; ++c) {
    const int kk0 = c * 128;
    if (c < 15) __builtin_prefetch(&Vp[(size_t)(kk0 + 128) * 64], 0, 0);
    // Stage V chunk transposed: sKV[d * 128 + key_local] (bf16, no cvt).
#pragma unroll
    for (int i = 0; i < 4; ++i) {
      int idx = tid + 256 * i, row = idx >> 3, c8 = (idx & 7) * 8;
      Pack8 p;
      p.q = *(const u32x4*)&Vp[(size_t)(kk0 + row) * 64 + c8];
#pragma unroll
      for (int e = 0; e < 8; ++e)
        sKV[(c8 + e) * 128 + row] = p.u[e];
    }
    __syncthreads();
#pragma unroll
    for (int ks = 0; ks < 2; ++ks) {
      int kb = kk0 + khalf * 64 + ks * 32;      // key base for this k-step
      Frag ap = fragA_f32(&sS[ln * 2048 + kb], g);   // probs fp32 -> bf16 frag
      Frag bv = fragB_bf16(
          &sKV[(ntile * 16 + ln) * 128 + khalf * 64 + ks * 32 + 16 * g]);
      accc = wmma_bf16(ap, bv, accc);
    }
    __syncthreads();
  }
  // Cross-wave (key-half) reduction into sCtx via LDS float atomics.
#pragma unroll
  for (int r = 0; r < 8; ++r)
    atomicAdd(&sCtx[(8 * g + r) * 64 + ntile * 16 + ln], accc[r]);
  __syncthreads();

  // Write ctx [B, S, 1024] with column h*64 + d.
#pragma unroll
  for (int i = 0; i < 4; ++i) {
    int idx = tid + 256 * i;
    int row = idx >> 6, d = idx & 63;
    ctx[((size_t)(b * 2048 + q0 + row)) * 1024 + h * 64 + d] = sCtx[row * 64 + d];
  }
}

// ---------------------------------------------------------------------------
extern "C" void kernel_launch(void* const* d_in, const int* in_sizes, int n_in,
                              void* d_out, int out_size, void* d_ws, size_t ws_size,
                              hipStream_t stream) {
  (void)in_sizes; (void)n_in; (void)out_size; (void)ws_size;
  const float* hs   = (const float*)d_in[0];
  const float* mask = (const float*)d_in[1];
  const float* Wq   = (const float*)d_in[2];
  const float* bq   = (const float*)d_in[3];
  const float* Wk   = (const float*)d_in[4];
  const float* bk   = (const float*)d_in[5];
  const float* Wv   = (const float*)d_in[6];
  const float* bv   = (const float*)d_in[7];

  float* ctx   = (float*)d_out;
  float* probs = ctx + (size_t)4 * 2048 * 1024;

  const size_t per = (size_t)4 * 16 * 2048 * 64;  // 8,388,608 elems each (bf16)
  unsigned short* Q = (unsigned short*)d_ws;
  unsigned short* K = Q + per;
  unsigned short* V = K + per;

  dim3 g1(16, 64, 3), b1(256);
  qkv_proj_kernel<<<g1, b1, 0, stream>>>(hs, Wq, bq, Wk, bk, Wv, bv, Q, K, V);

  dim3 g2(128, 16, 4), b2(256);
  attn_kernel<<<g2, b2, 0, stream>>>(Q, K, V, mask, ctx, probs);
}
